// GCPMessagePassing_37847251812665
// MI455X (gfx1250) — compile-verified, matrix operand order checked by reading, not run on hardware
//
#include <hip/hip_runtime.h>
#include <math.h>

#define N_NODES 10000
#define N_EDGES 160000
#define MSG 176           // 128 scalar + 48 vector
#define ETILE 16

typedef __attribute__((ext_vector_type(16))) _Float16 v16h;
typedef __attribute__((ext_vector_type(8)))  _Float16 v8h;
typedef __attribute__((ext_vector_type(8)))  float    v8f;

// ---- WMMA helpers -----------------------------------------------------------

__device__ __forceinline__ v8f wmma16(v16h a, v16h b, v8f c) {
  // D = A(16x32 f16) * B(32x16 f16) + C(16x16 f32)
  return __builtin_amdgcn_wmma_f32_16x16x32_f16(
      /*neg_a=*/false, a, /*neg_b=*/false, b,
      /*c_mod=*/(short)0, c, /*reuse_a=*/false, /*reuse_b=*/false);
}

// Load a 16x32 fragment (A-style). For A: row = M (edge). For B: pass the
// N-major (transposed) weight matrix; row = N. `stride` in halves, must be
// a multiple of 8 (16B). k0 multiple of 32.
__device__ __forceinline__ v16h load_frag(const _Float16* base, int stride,
                                          int k0, int lane) {
  const int row = lane & 15;
  const int hs  = lane >> 4;              // lane-half selects K offset +8/+24
  const _Float16* p = base + row * stride + k0 + hs * 8;
  v8h r0 = *(const v8h*)p;                // K = k0+hs*8 .. +7
  v8h r1 = *(const v8h*)(p + 16);         // K = k0+16+hs*8 .. +7
  v16h a;
#pragma unroll
  for (int i = 0; i < 8; ++i) { a[i] = r0[i]; a[i + 8] = r1[i]; }
  return a;
}

// ---- small utility kernels --------------------------------------------------

__global__ void zero_f32(float* p, int n) {
  int i = blockIdx.x * blockDim.x + threadIdx.x;
  if (i < n) p[i] = 0.0f;
}

// Wt[n*Kp + k] = (k<K) ? W[k*N + n] : 0   (f32 -> f16, transpose, K-pad)
__global__ void convertWT(const float* __restrict__ W, _Float16* __restrict__ Wt,
                          int K, int N, int Kp) {
  int i = blockIdx.x * blockDim.x + threadIdx.x;
  if (i >= N * Kp) return;
  int n = i / Kp, k = i % Kp;
  Wt[i] = (k < K) ? (_Float16)W[k * N + n] : (_Float16)0.0f;
}

__global__ void finalize_mean(const float* __restrict__ agg,
                              const float* __restrict__ cnt,
                              float* __restrict__ out) {
  int i = blockIdx.x * blockDim.x + threadIdx.x;
  if (i >= N_NODES * MSG) return;
  int n = i / MSG, j = i % MSG;
  float c = fmaxf(cnt[n], 1.0f);
  float v = agg[i] / c;
  if (j < 128) out[n * 128 + j] = v;
  else         out[N_NODES * 128 + n * 48 + (j - 128)] = v;
}

// ---- fused per-edge pipeline ------------------------------------------------

__global__ __launch_bounds__(128)
void gcp_main(const float* __restrict__ node_s, const float* __restrict__ node_v,
              const float* __restrict__ edge_s, const float* __restrict__ edge_v,
              const float* __restrict__ frames,
              const float* __restrict__ Wvd0, const float* __restrict__ bs0,
              const float* __restrict__ Wvu0, const float* __restrict__ bg0,
              const float* __restrict__ Wvd_m, const float* __restrict__ bs_m,
              const float* __restrict__ Wvu_m, const float* __restrict__ bg_m,
              const float* __restrict__ Wvd3, const float* __restrict__ bs3,
              const float* __restrict__ Wvu3, const float* __restrict__ bg3,
              const int* __restrict__ edge_index,
              const _Float16* __restrict__ Ws0t,   // [128][448]
              const _Float16* __restrict__ Wsmt,   // [2][128][192]
              const _Float16* __restrict__ Ws3t,   // [128][192]
              const _Float16* __restrict__ Wg0t,   // [16][128]
              const _Float16* __restrict__ Wgmt,   // [2][16][128]
              const _Float16* __restrict__ Wg3t,   // [16][128]
              float* __restrict__ agg, float* __restrict__ cnt) {
  __shared__ int   ridx[16], cidx[16];
  __shared__ float vmsg[16][108];     // [e][m*3+c], m<36 (layer 0 only)
  __shared__ float frm[16][9];        // [e][d*3+c]
  __shared__ float vh[16][108];       // [e][h*3+c] (36 for L0, 16 later)
  __shared__ float vbuf[16][48];      // running v  [e][m*3+c]
  __shared__ float vu[16][48];
  __shared__ float sbuf[16][128];     // running s (fp32)
  __shared__ float gate[16][16];
  __shared__ __align__(16) _Float16 scat[16][456];   // A operand (cat, f16)
  __shared__ __align__(16) _Float16 dsf16[16][136];  // gate GEMM A operand

  const int tid  = threadIdx.x;
  const int wid  = tid >> 5;
  const int lane = tid & 31;
  const int e0   = blockIdx.x * ETILE;

  if (tid < 16) {
    ridx[tid] = edge_index[e0 + tid];
    cidx[tid] = edge_index[N_EDGES + e0 + tid];
  }
  __syncthreads();

  // ---- gather: v_msg, frames, s part of cat ----
  for (int i = tid; i < 16 * 108; i += 128) {
    int e = i / 108, r = i % 108, m = r / 3, c = r % 3;
    float v;
    if (m < 16)      v = node_v[ridx[e] * 48 + m * 3 + c];
    else if (m < 20) v = edge_v[(size_t)(e0 + e) * 12 + (m - 16) * 3 + c];
    else             v = node_v[cidx[e] * 48 + (m - 20) * 3 + c];
    vmsg[e][r] = v;
  }
  for (int i = tid; i < 16 * 9; i += 128) frm[i / 9][i % 9] = frames[(size_t)e0 * 9 + i];
  for (int i = tid; i < 16 * 288; i += 128) {
    int e = i / 288, j = i % 288;
    float v;
    if (j < 128)      v = node_s[(size_t)ridx[e] * 128 + j];
    else if (j < 160) v = edge_s[(size_t)(e0 + e) * 32 + (j - 128)];
    else              v = node_s[(size_t)cidx[e] * 128 + (j - 160)];
    scat[e][36 + j] = (_Float16)v;
  }
  for (int i = tid; i < 16 * 16; i += 128)
    scat[i / 16][432 + (i % 16)] = (_Float16)0.0f;   // K padding 432..447
  __syncthreads();

  // ---- layer 0 vector down-projection + invariants (VALU, fp32) ----
  for (int i = tid; i < 16 * 36; i += 128) {
    int e = i / 36, h = i % 36;
    float a0 = 0.f, a1 = 0.f, a2 = 0.f;
    for (int m = 0; m < 36; ++m) {
      float w = Wvd0[m * 36 + h];
      a0 += vmsg[e][m * 3 + 0] * w;
      a1 += vmsg[e][m * 3 + 1] * w;
      a2 += vmsg[e][m * 3 + 2] * w;
    }
    vh[e][h * 3 + 0] = a0; vh[e][h * 3 + 1] = a1; vh[e][h * 3 + 2] = a2;
    scat[e][h] = (_Float16)sqrtf(a0 * a0 + a1 * a1 + a2 * a2 + 1e-8f);
#pragma unroll
    for (int d = 0; d < 3; ++d)
      scat[e][324 + h * 3 + d] =
          (_Float16)(a0 * frm[e][d * 3] + a1 * frm[e][d * 3 + 1] + a2 * frm[e][d * 3 + 2]);
  }
  __syncthreads();

  // ---- layer 0 scalar GEMM: [16,448] @ [448,128], each wave does 2 N-tiles
  {
    v8f acc[2] = {{}, {}};
    const int n0 = wid * 32;
    for (int kt = 0; kt < 14; ++kt) {
      v16h a  = load_frag(&scat[0][0], 456, kt * 32, lane);
      v16h b0 = load_frag(Ws0t + (size_t)n0 * 448, 448, kt * 32, lane);
      v16h b1 = load_frag(Ws0t + (size_t)(n0 + 16) * 448, 448, kt * 32, lane);
      acc[0] = wmma16(a, b0, acc[0]);
      acc[1] = wmma16(a, b1, acc[1]);
    }
    const int ncol = lane & 15, hs = lane >> 4;
#pragma unroll
    for (int t = 0; t < 2; ++t) {
      int N = n0 + t * 16 + ncol;
      float b = bs0[N];
#pragma unroll
      for (int r = 0; r < 8; ++r) {
        int M = r + hs * 8;
        float x = acc[t][r] + b;
        float y = x / (1.0f + __expf(-x));   // silu
        sbuf[M][N]  = y;
        dsf16[M][N] = (_Float16)y;
      }
    }
  }
  __syncthreads();

  // ---- layer 0 gate GEMM (wave 0) || vu (waves 1-3) ----
  if (wid == 0) {
    v8f g = {};
    for (int kt = 0; kt < 4; ++kt) {
      v16h a = load_frag(&dsf16[0][0], 136, kt * 32, lane);
      v16h b = load_frag(Wg0t, 128, kt * 32, lane);
      g = wmma16(a, b, g);
    }
    const int m = lane & 15, hs = lane >> 4;
    float bb = bg0[m];
#pragma unroll
    for (int r = 0; r < 8; ++r) {
      int e = r + hs * 8;
      float x = g[r] + bb;
      gate[e][m] = 1.0f / (1.0f + __expf(-x));
    }
  } else {
    for (int i = tid - 32; i < 16 * 48; i += 96) {
      int e = i / 48, rr = i % 48, m = rr / 3, c = rr % 3;
      float a = 0.f;
      for (int h = 0; h < 36; ++h) a += vh[e][h * 3 + c] * Wvu0[h * 16 + m];
      vu[e][rr] = a;
    }
  }
  __syncthreads();
  for (int i = tid; i < 16 * 48; i += 128) {
    int e = i / 48, rr = i % 48, m = rr / 3;
    vbuf[e][rr] = vu[e][rr] * gate[e][m];
  }
  __syncthreads();

  // ---- layers 1,2 (residual, silu/sigmoid) and layer 3 (residual, linear)
  for (int L = 0; L < 3; ++L) {
    const float* Wvd = (L < 2) ? Wvd_m + L * 256 : Wvd3;
    const float* Wvu = (L < 2) ? Wvu_m + L * 256 : Wvu3;
    const float* bsv = (L < 2) ? bs_m + L * 128  : bs3;
    const float* bgv = (L < 2) ? bg_m + L * 16   : bg3;
    const _Float16* Wst = (L < 2) ? Wsmt + (size_t)L * 128 * 192 : Ws3t;
    const _Float16* Wgt = (L < 2) ? Wgmt + (size_t)L * 16 * 128  : Wg3t;
    const bool act = (L < 2);

    // cat build: vh, vnorm, s(f16), scal
    for (int i = tid; i < 16 * 16; i += 128) {
      int e = i / 16, h = i % 16;
      float a0 = 0.f, a1 = 0.f, a2 = 0.f;
      for (int m = 0; m < 16; ++m) {
        float w = Wvd[m * 16 + h];
        a0 += vbuf[e][m * 3 + 0] * w;
        a1 += vbuf[e][m * 3 + 1] * w;
        a2 += vbuf[e][m * 3 + 2] * w;
      }
      vh[e][h * 3 + 0] = a0; vh[e][h * 3 + 1] = a1; vh[e][h * 3 + 2] = a2;
      scat[e][h] = (_Float16)sqrtf(a0 * a0 + a1 * a1 + a2 * a2 + 1e-8f);
#pragma unroll
      for (int d = 0; d < 3; ++d)
        scat[e][144 + h * 3 + d] =
            (_Float16)(a0 * frm[e][d * 3] + a1 * frm[e][d * 3 + 1] + a2 * frm[e][d * 3 + 2]);
    }
    for (int i = tid; i < 16 * 128; i += 128) {
      int e = i / 128, j = i % 128;
      scat[e][16 + j] = (_Float16)sbuf[e][j];
    }
    __syncthreads();

    // Ws GEMM: [16,192] @ [192,128]
    {
      v8f acc[2] = {{}, {}};
      const int n0 = wid * 32;
      for (int kt = 0; kt < 6; ++kt) {
        v16h a  = load_frag(&scat[0][0], 456, kt * 32, lane);
        v16h b0 = load_frag(Wst + (size_t)n0 * 192, 192, kt * 32, lane);
        v16h b1 = load_frag(Wst + (size_t)(n0 + 16) * 192, 192, kt * 32, lane);
        acc[0] = wmma16(a, b0, acc[0]);
        acc[1] = wmma16(a, b1, acc[1]);
      }
      const int ncol = lane & 15, hs = lane >> 4;
#pragma unroll
      for (int t = 0; t < 2; ++t) {
        int N = n0 + t * 16 + ncol;
        float b = bsv[N];
#pragma unroll
        for (int r = 0; r < 8; ++r) {
          int M = r + hs * 8;
          float x = acc[t][r] + b;
          float y = act ? (x / (1.0f + __expf(-x))) : x;
          sbuf[M][N] += y;             // residual
          dsf16[M][N] = (_Float16)y;   // gate input is ds (post-act)
        }
      }
    }
    __syncthreads();

    // gate GEMM (wave 0) || vu (waves 1-3)
    if (wid == 0) {
      v8f g = {};
      for (int kt = 0; kt < 4; ++kt) {
        v16h a = load_frag(&dsf16[0][0], 136, kt * 32, lane);
        v16h b = load_frag(Wgt, 128, kt * 32, lane);
        g = wmma16(a, b, g);
      }
      const int m = lane & 15, hs = lane >> 4;
      float bb = bgv[m];
#pragma unroll
      for (int r = 0; r < 8; ++r) {
        int e = r + hs * 8;
        float x = g[r] + bb;
        gate[e][m] = act ? (1.0f / (1.0f + __expf(-x))) : x;
      }
    } else {
      for (int i = tid - 32; i < 16 * 48; i += 96) {
        int e = i / 48, rr = i % 48, m = rr / 3, c = rr % 3;
        float a = 0.f;
        for (int h = 0; h < 16; ++h) a += vh[e][h * 3 + c] * Wvu[h * 16 + m];
        vu[e][rr] = a;
      }
    }
    __syncthreads();
    for (int i = tid; i < 16 * 48; i += 128) {
      int e = i / 48, rr = i % 48, m = rr / 3;
      vbuf[e][rr] += vu[e][rr] * gate[e][m];
    }
    __syncthreads();
  }

  // ---- scatter-mean accumulation ----
  for (int i = tid; i < 16 * MSG; i += 128) {
    int e = i / MSG, j = i % MSG;
    float v = (j < 128) ? sbuf[e][j] : vbuf[e][j - 128];
    atomicAdd(&agg[(size_t)cidx[e] * MSG + j], v);
  }
  if (tid < 16) atomicAdd(&cnt[cidx[tid]], 1.0f);
}

// ---- launcher ---------------------------------------------------------------

extern "C" void kernel_launch(void* const* d_in, const int* in_sizes, int n_in,
                              void* d_out, int out_size, void* d_ws, size_t ws_size,
                              hipStream_t stream) {
  (void)in_sizes; (void)n_in; (void)out_size; (void)ws_size;
  const float* node_s = (const float*)d_in[0];
  const float* node_v = (const float*)d_in[1];
  const float* edge_s = (const float*)d_in[2];
  const float* edge_v = (const float*)d_in[3];
  const float* frames = (const float*)d_in[4];
  const float* Wvd0   = (const float*)d_in[5];
  const float* Ws0    = (const float*)d_in[6];
  const float* bs0    = (const float*)d_in[7];
  const float* Wvu0   = (const float*)d_in[8];
  const float* Wg0    = (const float*)d_in[9];
  const float* bg0    = (const float*)d_in[10];
  const float* Wvd_m  = (const float*)d_in[11];
  const float* Ws_m   = (const float*)d_in[12];
  const float* bs_m   = (const float*)d_in[13];
  const float* Wvu_m  = (const float*)d_in[14];
  const float* Wg_m   = (const float*)d_in[15];
  const float* bg_m   = (const float*)d_in[16];
  const float* Wvd3   = (const float*)d_in[17];
  const float* Ws3    = (const float*)d_in[18];
  const float* bs3    = (const float*)d_in[19];
  const float* Wvu3   = (const float*)d_in[20];
  const float* Wg3    = (const float*)d_in[21];
  const float* bg3    = (const float*)d_in[22];
  const int* edge_index = (const int*)d_in[23];

  // workspace layout
  float* agg = (float*)d_ws;                       // [N_NODES][176]
  float* cnt = agg + (size_t)N_NODES * MSG;        // [N_NODES]
  _Float16* wb   = (_Float16*)(cnt + N_NODES);     // 16B-aligned (offset 7,080,000)
  _Float16* Ws0t = wb;                             // 128*448
  _Float16* Wsmt = Ws0t + 128 * 448;               // 2 * 128*192
  _Float16* Ws3t = Wsmt + 2 * 128 * 192;           // 128*192
  _Float16* Wg0t = Ws3t + 128 * 192;               // 16*128
  _Float16* Wgmt = Wg0t + 16 * 128;                // 2 * 16*128
  _Float16* Wg3t = Wgmt + 2 * 16 * 128;            // 16*128

  const int nz = N_NODES * MSG + N_NODES;
  zero_f32<<<(nz + 255) / 256, 256, 0, stream>>>(agg, nz);

  convertWT<<<(128 * 448 + 255) / 256, 256, 0, stream>>>(Ws0, Ws0t, 432, 128, 448);
  convertWT<<<(128 * 192 + 255) / 256, 256, 0, stream>>>(Ws_m, Wsmt, 192, 128, 192);
  convertWT<<<(128 * 192 + 255) / 256, 256, 0, stream>>>(Ws_m + 192 * 128, Wsmt + 128 * 192, 192, 128, 192);
  convertWT<<<(128 * 192 + 255) / 256, 256, 0, stream>>>(Ws3, Ws3t, 192, 128, 192);
  convertWT<<<(16 * 128 + 255) / 256, 256, 0, stream>>>(Wg0, Wg0t, 128, 16, 128);
  convertWT<<<(16 * 128 + 255) / 256, 256, 0, stream>>>(Wg_m, Wgmt, 128, 16, 128);
  convertWT<<<(16 * 128 + 255) / 256, 256, 0, stream>>>(Wg_m + 128 * 16, Wgmt + 16 * 128, 128, 16, 128);
  convertWT<<<(16 * 128 + 255) / 256, 256, 0, stream>>>(Wg3, Wg3t, 128, 16, 128);

  gcp_main<<<N_EDGES / ETILE, 128, 0, stream>>>(
      node_s, node_v, edge_s, edge_v, frames,
      Wvd0, bs0, Wvu0, bg0,
      Wvd_m, bs_m, Wvu_m, bg_m,
      Wvd3, bs3, Wvu3, bg3,
      edge_index,
      Ws0t, Wsmt, Ws3t, Wg0t, Wgmt, Wg3t,
      agg, cnt);

  finalize_mean<<<(N_NODES * MSG + 255) / 256, 256, 0, stream>>>(agg, cnt, (float*)d_out);
}
